// QCNNModel_57810259804661
// MI455X (gfx1250) — matrix operand order repaired and checked by reading, not compiled, not run on hardware
//
#include <hip/hip_runtime.h>
#include <hip/hip_bf16.h>

typedef __attribute__((ext_vector_type(16))) _Float16 v16h;
typedef __attribute__((ext_vector_type(8)))  _Float16 v8h;
typedef __attribute__((ext_vector_type(8)))  float    v8f;

// ---------------- workspace layout (units: _Float16 elements) ----------------
#define OFF_QW1 0        // [16][2*2]   = 64
#define OFF_QW2 64       // [16][16*3]  = 768
#define OFF_QW3 832      // [32][16*5]  = 2560
#define OFF_QW4 3392     // [32][32*5]  = 5120
#define OFF_QWD 8512     // [64][864]   = 55296
#define OFF_QWO 63808    // [64]
#define WS_HALFS 63872   // ~128 KB of f16 in d_ws

// -------- LDS layout: tap-padded weights, biases, per-wave activation bufs ---
// padded weights (row stride = KPAD halfs):
#define LW_QW1 0                    // 16 x 32   = 512
#define LW_QW2 512                  // 16 x 64   = 1024
#define LW_QW3 1536                 // 32 x 128  = 4096
#define LW_QW4 5632                 // 32 x 256  = 8192
#define WQP_H  13824                // total padded weight halfs (27,648 B)
#define BIAS_H 352                  // 176 floats as halfs (4B aligned region)
// biases inside the float view: b1@0 b2@16 b3@32 b4@64 bd@96 bo@160
#define XBUF_H 1152                 // [8][144] conv1 input (rows 2..7 zero)
#define BUF_H  2560                 // max(16*144, 32*80)
#define FLAT_H (16*864)             // 13824
#define H64_H  2048                 // float[16][64] viewed as halfs
#define WAVE_H (XBUF_H + 2*BUF_H + FLAT_H + H64_H)   // 22144
#define NWAVES 2
#define SMEM_HALFS (WQP_H + BIAS_H + NWAVES*WAVE_H)  // 58464
#define SMEM_BYTES ((size_t)SMEM_HALFS * 2)          // 116,928 B

__device__ __forceinline__ float lrelu(float v) { return v > 0.0f ? v : 0.1f * v; }

// ---------------------------------------------------------------------------
// Kernel 0: per-tensor symmetric int8 quantize-dequantize (Brevitas semantics)
// ---------------------------------------------------------------------------
__global__ __launch_bounds__(256) void quant_kernel(
    const float* __restrict__ w1, const float* __restrict__ w2,
    const float* __restrict__ w3, const float* __restrict__ w4,
    const float* __restrict__ wd, const float* __restrict__ wo,
    _Float16* __restrict__ ws)
{
    __shared__ float red[256];
    const float* src; _Float16* dst; int nel;
    switch (blockIdx.x) {
        case 0:  src = w1; dst = ws + OFF_QW1; nel = 64;    break;
        case 1:  src = w2; dst = ws + OFF_QW2; nel = 768;   break;
        case 2:  src = w3; dst = ws + OFF_QW3; nel = 2560;  break;
        case 3:  src = w4; dst = ws + OFF_QW4; nel = 5120;  break;
        case 4:  src = wd; dst = ws + OFF_QWD; nel = 55296; break;
        default: src = wo; dst = ws + OFF_QWO; nel = 64;    break;
    }
    float m = 0.0f;
    for (int i = threadIdx.x; i < nel; i += 256) m = fmaxf(m, fabsf(src[i]));
    red[threadIdx.x] = m;
    __syncthreads();
    for (int st = 128; st > 0; st >>= 1) {
        if (threadIdx.x < st)
            red[threadIdx.x] = fmaxf(red[threadIdx.x], red[threadIdx.x + st]);
        __syncthreads();
    }
    const float s   = red[0] * (1.0f / 127.0f);
    const float inv = 1.0f / s;
    for (int i = threadIdx.x; i < nel; i += 256) {
        float q = rintf(src[i] * inv);                 // RTE == jnp.round
        q = fminf(127.0f, fmaxf(-127.0f, q));
        dst[i] = (_Float16)(q * s);
    }
}

// ---------------------------------------------------------------------------
// Stage a conv weight tensor [COUT][CIN*KW] -> LDS tap-padded [COUT][KPAD]
// (k = ic*KWP + kw; slots with kw>=KW or ic>=CIN are zero). Branch-free:
// loads use clamped indices, the pad selects 0 afterwards.
// ---------------------------------------------------------------------------
template<int CIN, int COUT, int KW, int KWP, int KPAD>
__device__ void stage_w(const _Float16* __restrict__ src,
                        _Float16* __restrict__ dst, int tid, int nthr)
{
    for (int i = tid; i < COUT * KPAD; i += nthr) {
        int oc = i / KPAD, k = i - oc * KPAD;
        int ic = k / KWP, kw = k & (KWP - 1);
        int ics = ic < CIN ? ic : CIN - 1;
        int kws = kw < KW  ? kw : KW - 1;
        _Float16 v = src[oc * (CIN * KW) + ics * KW + kws];
        dst[i] = (ic < CIN && kw < KW) ? v : (_Float16)0.0f;
    }
}

// ---------------------------------------------------------------------------
// Conv layer as WMMA GEMM, one wave / one sample.
//   M = COUT, K = CIN*KWP (tap-padded, pow2), N = 16 output positions/tile.
// A (16x32, contiguous k order chosen here): lane l: M=l&15, k = hi*16 + e
//   -> two aligned 16B LDS loads from the padded weight row.
// B (KxN layout): lane l: N=l&15, k = hi*16 + e -> statically structured
//   gather: 16/KWP channel groups, KWP contiguous halfs each. Unconditional.
// ---------------------------------------------------------------------------
template<int CIN, int COUT, int KW, int KWP, int LOUT, int ISTR, int OSTR>
__device__ void conv_wmma(const _Float16* __restrict__ in,
                          _Float16* __restrict__ out,
                          const _Float16* __restrict__ w,    // LDS [COUT][KPAD]
                          const float* __restrict__ bias,    // LDS [COUT]
                          int lane)
{
    constexpr int S      = (KWP == 4) ? 2 : 3;
    constexpr int KPAD   = ((CIN * KWP) + 31) & ~31;
    constexpr int KSTEPS = KPAD / 32;
    constexpr int NT     = (LOUT + 15) / 16;
    constexpr int MT     = COUT / 16;
    constexpr int G      = 16 / KWP;          // channel groups per 16-k run
    const int n   = lane & 15;
    const int hi  = lane >> 4;
    const int klo = hi * 16;

    for (int mt = 0; mt < MT; ++mt) {
        v16h afr[KSTEPS];
        const int am = mt * 16 + n;
        #pragma unroll
        for (int kk = 0; kk < KSTEPS; ++kk) {
            const v8h* ap = (const v8h*)(w + am * KPAD + kk * 32 + klo);
            afr[kk] = __builtin_shufflevector(ap[0], ap[1],
                        0,1,2,3,4,5,6,7,8,9,10,11,12,13,14,15);
        }
        for (int nt = 0; nt < NT; ++nt) {
            v8f acc = {0.f,0.f,0.f,0.f,0.f,0.f,0.f,0.f};
            const _Float16* bin = in + nt * 16 + n;
            #pragma unroll
            for (int kk = 0; kk < KSTEPS; ++kk) {
                const int icb = (kk * 32 + hi * 16) >> S;
                v16h bfr;
                #pragma unroll
                for (int g = 0; g < G; ++g) {
                    const _Float16* r = bin + (icb + g) * ISTR;
                    #pragma unroll
                    for (int kw = 0; kw < KWP; ++kw)
                        bfr[g * KWP + kw] = r[kw];
                }
                acc = __builtin_amdgcn_wmma_f32_16x16x32_f16(
                          false, afr[kk], false, bfr, (short)0, acc, false, false);
            }
            #pragma unroll
            for (int r = 0; r < 8; ++r) {
                int m = mt * 16 + hi * 8 + r;
                int p = nt * 16 + n;
                if (p < LOUT)
                    out[m * OSTR + p] = (_Float16)lrelu(acc[r] + bias[m]);
            }
        }
    }
}

template<int C, int LOUT, int STR>
__device__ void zero_pad(_Float16* __restrict__ buf, int lane)
{
    constexpr int PAD = STR - LOUT;
    for (int i = lane; i < C * PAD; i += 32) {
        int c = i / PAD, j = i - c * PAD;
        buf[c * STR + LOUT + j] = (_Float16)0.0f;
    }
}

template<int C, int LOUT, int ISTR, int OSTR>
__device__ void pool2x(const _Float16* __restrict__ in,
                       _Float16* __restrict__ out, int lane)
{
    for (int i = lane; i < C * LOUT; i += 32) {
        int c = i / LOUT, j = i - c * LOUT;
        float a = (float)in[c * ISTR + 2 * j];
        float b = (float)in[c * ISTR + 2 * j + 1];
        out[c * OSTR + j] = (_Float16)fmaxf(a, b);
    }
}

// ---------------------------------------------------------------------------
// Kernel 1: fully fused QCNN forward. One wave <-> 16 samples through the conv
// stack (LDS-resident activations), then dense 864->64 as a dense N=16 WMMA
// GEMM (864 = 27*32 exactly) and a per-lane 64->1 head.
// ---------------------------------------------------------------------------
__global__ __launch_bounds__(64) void qcnn_fused(
    const float* __restrict__ x,
    const float* __restrict__ b1, const float* __restrict__ b2,
    const float* __restrict__ b3, const float* __restrict__ b4,
    const float* __restrict__ bd, const float* __restrict__ bo,
    const _Float16* __restrict__ qw,
    float* __restrict__ out, int nsamp)
{
    extern __shared__ _Float16 smem[];
    const int lane = threadIdx.x & 31;
    const int wave = threadIdx.x >> 5;
    const int tid  = threadIdx.x;

    // ---- stage tap-padded conv weights + biases into LDS (one time) ----
    stage_w< 2, 16, 2, 4,  32>(qw + OFF_QW1, smem + LW_QW1, tid, blockDim.x);
    stage_w<16, 16, 3, 4,  64>(qw + OFF_QW2, smem + LW_QW2, tid, blockDim.x);
    stage_w<16, 32, 5, 8, 128>(qw + OFF_QW3, smem + LW_QW3, tid, blockDim.x);
    stage_w<32, 32, 5, 8, 256>(qw + OFF_QW4, smem + LW_QW4, tid, blockDim.x);
    float* bsh = (float*)(smem + WQP_H);
    for (int i = tid; i < 161; i += blockDim.x) {
        float v;
        if      (i <  16) v = b1[i];
        else if (i <  32) v = b2[i - 16];
        else if (i <  64) v = b3[i - 32];
        else if (i <  96) v = b4[i - 64];
        else if (i < 160) v = bd[i - 96];
        else              v = bo[0];
        bsh[i] = v;
    }
    __syncthreads();

    const _Float16* qw1 = smem + LW_QW1;
    const _Float16* qw2 = smem + LW_QW2;
    const _Float16* qw3 = smem + LW_QW3;
    const _Float16* qw4 = smem + LW_QW4;

    _Float16* pw   = smem + WQP_H + BIAS_H + wave * WAVE_H;
    _Float16* xbuf = pw;                            // [8][144]
    _Float16* bufA = pw + XBUF_H;
    _Float16* bufB = bufA + BUF_H;
    _Float16* flat = bufB + BUF_H;                  // [16][864] f16
    float*    h64  = (float*)(flat + FLAT_H);       // [16][64]  f32

    const int gw = blockIdx.x * NWAVES + wave;
    const int s0 = gw * 16;
    if (s0 >= nsamp) return;                        // wave-uniform guard

    // xbuf rows 2..7 are zero forever (conv1 tap-pad reads land there)
    for (int i = lane; i < 6 * 144; i += 32) xbuf[2 * 144 + i] = (_Float16)0.0f;

    for (int si = 0; si < 16; ++si) {
        // ---- stage x[s]: [2][128] f32 -> f16, rows padded to 144 ----
        const float* xs = x + (size_t)(s0 + si) * 256;
        for (int i = lane; i < 2 * 144; i += 32) {
            int c = i / 144, j = i - c * 144;
            int js = j < 128 ? j : 127;
            float v = xs[c * 128 + js];
            xbuf[i] = (j < 128) ? (_Float16)v : (_Float16)0.0f;
        }
        conv_wmma< 2, 16, 2, 4, 127, 144, 144>(xbuf, bufA, qw1, bsh +  0, lane);
        zero_pad<16, 127, 144>(bufA, lane);
        conv_wmma<16, 16, 3, 4, 125, 144, 144>(bufA, bufB, qw2, bsh + 16, lane);
        zero_pad<16, 125, 144>(bufB, lane);
        pool2x<16, 62, 144, 80>(bufB, bufA, lane);
        zero_pad<16, 62, 80>(bufA, lane);
        conv_wmma<16, 32, 5, 8,  58,  80,  80>(bufA, bufB, qw3, bsh + 32, lane);
        zero_pad<32, 58, 80>(bufB, lane);
        conv_wmma<32, 32, 5, 8,  54,  80,  80>(bufB, bufA, qw4, bsh + 64, lane);
        // ---- pool2 + flatten straight into flat[si][c*27+j] ----
        _Float16* fr = flat + si * 864;
        for (int i = lane; i < 32 * 27; i += 32) {
            int c = i / 27, j = i - c * 27;
            float a = (float)bufA[c * 80 + 2 * j];
            float b = (float)bufA[c * 80 + 2 * j + 1];
            fr[i] = (_Float16)fmaxf(a, b);          // i == c*27 + j
        }
    }

    // ---- dense 864 -> 64, N = 16 samples, K = 864 = 27*32 exactly ----
    const _Float16* qwd = qw + OFF_QWD;             // 110 KB, lives in L2
    const int n = lane & 15, hi = lane >> 4, klo = hi * 16;
    for (int mt = 0; mt < 4; ++mt) {
        v8f acc = {0.f,0.f,0.f,0.f,0.f,0.f,0.f,0.f};
        const int am = mt * 16 + n;
        for (int kk = 0; kk < 27; ++kk) {
            const v8h* ap = (const v8h*)(qwd + (size_t)am * 864 + kk * 32 + klo);
            v16h a = __builtin_shufflevector(ap[0], ap[1],
                        0,1,2,3,4,5,6,7,8,9,10,11,12,13,14,15);
            const v8h* bp = (const v8h*)(flat + n * 864 + kk * 32 + hi * 16);
            v16h bfr = __builtin_shufflevector(bp[0], bp[1],
                        0,1,2,3,4,5,6,7,8,9,10,11,12,13,14,15);
            acc = __builtin_amdgcn_wmma_f32_16x16x32_f16(
                      false, a, false, bfr, (short)0, acc, false, false);
        }
        #pragma unroll
        for (int r = 0; r < 8; ++r) {
            int oc = mt * 16 + hi * 8 + r;
            h64[n * 64 + oc] = lrelu(acc[r] + bsh[96 + oc]);
        }
    }

    // ---- head: 64 -> 1 per sample ----
    const _Float16* qwo = qw + OFF_QWO;
    if (lane < 16) {
        float a = bsh[160];
        #pragma unroll 8
        for (int c = 0; c < 64; ++c) a += (float)qwo[c] * h64[lane * 64 + c];
        if (s0 + lane < nsamp) out[s0 + lane] = a;
    }
}

// ---------------------------------------------------------------------------
extern "C" void kernel_launch(void* const* d_in, const int* in_sizes, int n_in,
                              void* d_out, int out_size, void* d_ws, size_t ws_size,
                              hipStream_t stream)
{
    const float* x  = (const float*)d_in[0];
    const float* w1 = (const float*)d_in[1];
    const float* b1 = (const float*)d_in[2];
    const float* w2 = (const float*)d_in[3];
    const float* b2 = (const float*)d_in[4];
    const float* w3 = (const float*)d_in[5];
    const float* b3 = (const float*)d_in[6];
    const float* w4 = (const float*)d_in[7];
    const float* b4 = (const float*)d_in[8];
    const float* wd = (const float*)d_in[9];
    const float* bd = (const float*)d_in[10];
    const float* wo = (const float*)d_in[11];
    const float* bo = (const float*)d_in[12];
    _Float16* qw = (_Float16*)d_ws;
    float* out = (float*)d_out;

    quant_kernel<<<6, 256, 0, stream>>>(w1, w2, w3, w4, wd, wo, qw);

    const int nsamp = out_size;                     // B (x1 output)
    const int nwg   = (nsamp + 15) / 16;            // 16 samples per wave
    const int nblk  = (nwg + NWAVES - 1) / NWAVES;

    hipFuncSetAttribute((const void*)qcnn_fused,
                        hipFuncAttributeMaxDynamicSharedMemorySize,
                        (int)SMEM_BYTES);
    qcnn_fused<<<nblk, 32 * NWAVES, SMEM_BYTES, stream>>>(
        x, b1, b2, b3, b4, bd, bo, qw, out, nsamp);
}
